// OperationLayer_83623013253742
// MI455X (gfx1250) — compile-verified, matrix operand order checked by reading, not run on hardware
//
#include <hip/hip_runtime.h>

typedef __attribute__((ext_vector_type(16))) __bf16 v16bf;
typedef __attribute__((ext_vector_type(8)))  float  v8f;

#define KC   128            // K-chunk staged in LDS
#define KCP  (KC + 8)       // LDS row pitch in bf16 (16B pad -> 4-bank stride, conflict-free b128)

__device__ __forceinline__ float elu_f(float x) {
    return x > 0.f ? x : (__expf(x) - 1.f);
}

// ---------------- utility kernels ----------------

__global__ void k_zero_f32(float* __restrict__ p, long n) {
    long i = (long)blockIdx.x * blockDim.x + threadIdx.x;
    long s = (long)gridDim.x * blockDim.x;
    for (; i < n; i += s) p[i] = 0.f;
}

__global__ void k_cvt_bf16(const float* __restrict__ in, __bf16* __restrict__ out, long n) {
    long i = (long)blockIdx.x * blockDim.x + threadIdx.x;
    long s = (long)gridDim.x * blockDim.x;
    for (; i < n; i += s) out[i] = (__bf16)in[i];
}

// one wave32 per edge; lane loads float4 (32 lanes * 16B = 512B = one 128-wide row)
__global__ void k_scatter_add(const float* __restrict__ vals,
                              const int* __restrict__ srcIdx,
                              const int* __restrict__ dstIdx,
                              float* __restrict__ agg, int nEdges) {
    int e = (int)blockIdx.x * (blockDim.x >> 5) + (threadIdx.x >> 5);
    if (e >= nEdges) return;
    int lane = threadIdx.x & 31;
    int s = srcIdx[e];
    int d = dstIdx[e];
    const float4 v = ((const float4*)(vals + (long)s * 128))[lane];
    float* o = agg + (long)d * 128 + lane * 4;
    atomicAdd(o + 0, v.x);
    atomicAdd(o + 1, v.y);
    atomicAdd(o + 2, v.z);
    atomicAdd(o + 3, v.w);
}

__global__ void k_zero_edge_rows(float* __restrict__ out, int nRows, int cols) {
    int t = threadIdx.x;
    if (t < cols) {
        out[t] = 0.f;
        out[(long)(nRows - 1) * cols + t] = 0.f;
    }
}

// ---------------- WMMA GEMM: OUT[n,j] = act( sum_k X[n,k]*W[j,k] + bias[j] ) ----------------
// X: nRows x K bf16 row-major (ld=ldx); W: J x K bf16 row-major.
// Block: 256 threads = 8 waves; block tile = 256 rows x 64 cols; wave tile = 32 rows x 64 cols
// (8 f32 accumulators). B panel (64 cols x KC) double-buffered in LDS, shared by all waves;
// next chunk's global W loads are issued before the compute loop so they overlap WMMA work.
// A loads software-pipelined (prefetch next k-step into separate regs).
// Register layouts used here (contiguous k order, same for A and B; not the ISA table order):
//   A 16x32 bf16 : lane L holds row (L&15), K = (L>>4)*16..+15 (contiguous 32B)
//   B 32x16 bf16 : lane L holds col (L&15) (= one W row),  K = (L>>4)*16..+15 (contiguous 32B)
//   C/D 16x16 f32: lane L col = (L&15); VGPR v row = v + 8*(L>>4)
template <bool ELU_ACT, bool OUT_BF16>
__launch_bounds__(256)
__global__ void k_gemm_wmma(const __bf16* __restrict__ X, int ldx,
                            const __bf16* __restrict__ W,
                            const float* __restrict__ bias,
                            void* __restrict__ outPtr, int ldo, int colOff,
                            int nRows, int K) {
    __shared__ __bf16 ldsB[2][64 * KCP];   // 2 * 17408 B = 34 KB

    const int tid  = threadIdx.x;
    const int lane = tid & 31;
    const int wave = tid >> 5;
    const int rowBase = blockIdx.x * 256 + wave * 32;
    const int colBase = blockIdx.y * 64;
    const int lm = lane & 15;
    const int kh = (lane >> 4) << 4;

    int ar0 = rowBase + lm;       if (ar0 >= nRows) ar0 = nRows - 1;  // clamp loads
    int ar1 = rowBase + 16 + lm;  if (ar1 >= nRows) ar1 = nRows - 1;
    const __bf16* ap0 = X + (long)ar0 * ldx + kh;
    const __bf16* ap1 = X + (long)ar1 * ldx + kh;

    // B-panel staging helpers: 1024 16B units per chunk, 4 per thread, coalesced.
    auto wLoad = [&](int i, int kc) -> uint4 {
        int u  = tid * 4 + i;
        int j  = u >> 4;             // 0..63   (KC/8 = 16 units per row)
        int t8 = (u & 15) << 3;      // bf16 offset within row
        return *(const uint4*)(W + (long)(colBase + j) * K + kc + t8);
    };
    auto sStore = [&](int b, int i, uint4 v) {
        int u  = tid * 4 + i;
        int j  = u >> 4;
        int t8 = (u & 15) << 3;
        *(uint4*)&ldsB[b][j * KCP + t8] = v;
    };
    // B tile read from LDS as two 16B ds loads (16B-aligned by construction)
    union BV { uint4 u[2]; v16bf v; };
    auto bTile = [&](int b, int t, int ks) -> v16bf {
        BV r;
        const int idx = (t * 16 + lm) * KCP + ks + kh;
        r.u[0] = *(const uint4*)&ldsB[b][idx];
        r.u[1] = *(const uint4*)&ldsB[b][idx + 8];
        return r.v;
    };

    // stage chunk 0
    uint4 st0 = wLoad(0, 0), st1 = wLoad(1, 0), st2 = wLoad(2, 0), st3 = wLoad(3, 0);
    sStore(0, 0, st0); sStore(0, 1, st1); sStore(0, 2, st2); sStore(0, 3, st3);
    __syncthreads();

    v8f acc0 = {}, acc1 = {}, acc2 = {}, acc3 = {};
    v8f acc4 = {}, acc5 = {}, acc6 = {}, acc7 = {};

    v16bf a0 = *(const v16bf*)(ap0);
    v16bf a1 = *(const v16bf*)(ap1);

    const int nChunks = K / KC;
    for (int c = 0; c < nChunks; ++c) {
        const int kc = c * KC;
        const bool more = (c + 1 < nChunks);
        if (more) {   // issue next chunk's global loads; they complete under the WMMA work
            st0 = wLoad(0, kc + KC); st1 = wLoad(1, kc + KC);
            st2 = wLoad(2, kc + KC); st3 = wLoad(3, kc + KC);
        }
        const int b = c & 1;
#pragma unroll
        for (int ks = 0; ks < KC; ks += 32) {
            int kn = kc + ks + 32;
            if (kn >= K) kn = 0;                  // harmless wrap for tail prefetch
            v16bf a0n = *(const v16bf*)(ap0 + kn);
            v16bf a1n = *(const v16bf*)(ap1 + kn);
            v16bf b0 = bTile(b, 0, ks);
            v16bf b1 = bTile(b, 1, ks);
            v16bf b2 = bTile(b, 2, ks);
            v16bf b3 = bTile(b, 3, ks);
            acc0 = __builtin_amdgcn_wmma_f32_16x16x32_bf16(false, a0, false, b0, (short)0, acc0, false, false);
            acc1 = __builtin_amdgcn_wmma_f32_16x16x32_bf16(false, a0, false, b1, (short)0, acc1, false, false);
            acc2 = __builtin_amdgcn_wmma_f32_16x16x32_bf16(false, a0, false, b2, (short)0, acc2, false, false);
            acc3 = __builtin_amdgcn_wmma_f32_16x16x32_bf16(false, a0, false, b3, (short)0, acc3, false, false);
            acc4 = __builtin_amdgcn_wmma_f32_16x16x32_bf16(false, a1, false, b0, (short)0, acc4, false, false);
            acc5 = __builtin_amdgcn_wmma_f32_16x16x32_bf16(false, a1, false, b1, (short)0, acc5, false, false);
            acc6 = __builtin_amdgcn_wmma_f32_16x16x32_bf16(false, a1, false, b2, (short)0, acc6, false, false);
            acc7 = __builtin_amdgcn_wmma_f32_16x16x32_bf16(false, a1, false, b3, (short)0, acc7, false, false);
            a0 = a0n; a1 = a1n;
        }
        if (more) {
            __syncthreads();                       // everyone done reading buf b^1 (chunk c-1)
            sStore(b ^ 1, 0, st0); sStore(b ^ 1, 1, st1);
            sStore(b ^ 1, 2, st2); sStore(b ^ 1, 3, st3);
            __syncthreads();
        }
    }

    const float bv0 = bias[colBase +  0 + lm];
    const float bv1 = bias[colBase + 16 + lm];
    const float bv2 = bias[colBase + 32 + lm];
    const float bv3 = bias[colBase + 48 + lm];
    const int rOff = (lane >> 4) * 8;

    auto emit = [&](int rt, const v8f& A0, const v8f& A1, const v8f& A2, const v8f& A3) {
#pragma unroll
        for (int v = 0; v < 8; ++v) {
            int row = rowBase + rt * 16 + rOff + v;
            if (row >= nRows) continue;
            float r0 = A0[v] + bv0;
            float r1 = A1[v] + bv1;
            float r2 = A2[v] + bv2;
            float r3 = A3[v] + bv3;
            if (ELU_ACT) { r0 = elu_f(r0); r1 = elu_f(r1); r2 = elu_f(r2); r3 = elu_f(r3); }
            long o = (long)row * ldo + colOff + colBase + lm;
            if (OUT_BF16) {
                __bf16* out = (__bf16*)outPtr;
                out[o +  0] = (__bf16)r0;
                out[o + 16] = (__bf16)r1;
                out[o + 32] = (__bf16)r2;
                out[o + 48] = (__bf16)r3;
            } else {
                float* out = (float*)outPtr;
                out[o +  0] = r0;
                out[o + 16] = r1;
                out[o + 32] = r2;
                out[o + 48] = r3;
            }
        }
    };
    emit(0, acc0, acc1, acc2, acc3);
    emit(1, acc4, acc5, acc6, acc7);
}

// ---------------- host side ----------------

extern "C" void kernel_launch(void* const* d_in, const int* in_sizes, int n_in,
                              void* d_out, int out_size, void* d_ws, size_t ws_size,
                              hipStream_t stream) {
    const int D = 128, H = 256, OUTC = 128, CIN = 768;
    const int N = in_sizes[0] / D;

    const float* operations = (const float*)d_in[0];
    const float* related    = (const float*)d_in[1];
    const float* materials  = (const float*)d_in[2];
    const float* resources  = (const float*)d_in[3];
    const int* matE = (const int*)d_in[4]; const int eMat = in_sizes[4] / 2;
    const int* resE = (const int*)d_in[5]; const int eRes = in_sizes[5] / 2;
    const int* preE = (const int*)d_in[6]; const int ePre = in_sizes[6] / 2;

    // params dict order: self, items, materials, resources, predecessors, successors, combined
    const float* Wf[7][3];
    const float* Bf[7][3];
    for (int m = 0; m < 7; ++m)
        for (int l = 0; l < 3; ++l) {
            Wf[m][l] = (const float*)d_in[7 + m * 6 + l * 2];
            Bf[m][l] = (const float*)d_in[7 + m * 6 + l * 2 + 1];
        }

    // workspace layout
    char* ws = (char*)d_ws;
    size_t off = 0;
    auto take = [&](size_t bytes) -> char* {
        char* p = ws + off;
        off = (off + bytes + 255) & ~(size_t)255;
        return p;
    };
    __bf16* concat = (__bf16*)take((size_t)N * CIN * 2);
    __bf16* h1     = (__bf16*)take((size_t)N * H * 2);
    __bf16* h2     = (__bf16*)take((size_t)N * H * 2);
    __bf16* xb     = (__bf16*)take((size_t)N * D * 2);
    float*  agg    = (float*) take((size_t)N * D * 4);

    __bf16* Wb[7][3];
    long wN[7][3];
    for (int m = 0; m < 7; ++m) {
        wN[m][0] = (long)H * (m == 6 ? CIN : D);
        wN[m][1] = (long)H * H;
        wN[m][2] = (long)OUTC * H;
        for (int l = 0; l < 3; ++l)
            Wb[m][l] = (__bf16*)take((size_t)wN[m][l] * 2);
    }

    // convert all weights fp32 -> bf16
    for (int m = 0; m < 7; ++m)
        for (int l = 0; l < 3; ++l) {
            long n = wN[m][l];
            k_cvt_bf16<<<(int)((n + 255) / 256), 256, 0, stream>>>(Wf[m][l], Wb[m][l], n);
        }

    const long nD = (long)N * D;
    const int cvtBlocks = (int)((nD + 255) / 256);
    const int rowBlocks = (N + 255) / 256;
    const dim3 blk(256);

    auto gemm = [&](const __bf16* X, int ldx, int K, const __bf16* W, const float* b,
                    int J, void* out, int ldo, int colOff, bool eluAct, bool outBf16) {
        dim3 grid(rowBlocks, J / 64);
        if (eluAct)
            k_gemm_wmma<true, true><<<grid, blk, 0, stream>>>(X, ldx, W, b, out, ldo, colOff, N, K);
        else if (outBf16)
            k_gemm_wmma<false, true><<<grid, blk, 0, stream>>>(X, ldx, W, b, out, ldo, colOff, N, K);
        else
            k_gemm_wmma<false, false><<<grid, blk, 0, stream>>>(X, ldx, W, b, out, ldo, colOff, N, K);
    };

    auto mlp_branch = [&](const __bf16* xin, int m, int colOff) {
        gemm(xin, D, D, Wb[m][0], Bf[m][0], H, h1, H, 0, true, true);
        gemm(h1, H, H, Wb[m][1], Bf[m][1], H, h2, H, 0, true, true);
        gemm(h2, H, H, Wb[m][2], Bf[m][2], OUTC, concat, CIN, colOff, false, true);
    };

    auto scatter = [&](const float* vals, const int* src, const int* dst, int nE) {
        k_zero_f32<<<2048, 256, 0, stream>>>(agg, nD);
        k_scatter_add<<<(nE + 7) / 8, 256, 0, stream>>>(vals, src, dst, agg, nE);
        k_cvt_bf16<<<cvtBlocks, 256, 0, stream>>>(agg, xb, nD);
    };

    // concat column layout: [preds 0 | succs 128 | res 256 | mat 384 | item 512 | self 640]

    // self (m=0)
    k_cvt_bf16<<<cvtBlocks, 256, 0, stream>>>(operations, xb, nD);
    mlp_branch(xb, 0, 5 * OUTC);
    // items (m=1)
    k_cvt_bf16<<<cvtBlocks, 256, 0, stream>>>(related, xb, nD);
    mlp_branch(xb, 1, 4 * OUTC);
    // materials (m=2): gather materials[edge[1]] -> op edge[0]
    scatter(materials, matE + eMat, matE, eMat);
    mlp_branch(xb, 2, 3 * OUTC);
    // resources (m=3)
    scatter(resources, resE + eRes, resE, eRes);
    mlp_branch(xb, 3, 2 * OUTC);
    // predecessors (m=4): resources[prec[1]] -> prec[0]
    scatter(resources, preE + ePre, preE, ePre);
    mlp_branch(xb, 4, 0);
    // successors (m=5): resources[prec[0]] -> prec[1]
    scatter(resources, preE, preE + ePre, ePre);
    mlp_branch(xb, 5, OUTC);

    // combined MLP (m=6): 768 -> 256 -> 256 -> 128, final output fp32 to d_out
    gemm(concat, CIN, CIN, Wb[6][0], Bf[6][0], H, h1, H, 0, true, true);
    gemm(h1, H, H, Wb[6][1], Bf[6][1], H, h2, H, 0, true, true);
    gemm(h2, H, H, Wb[6][2], Bf[6][2], OUTC, d_out, OUTC, 0, false, false);

    // rows 0 and N-1 are zero in the reference output
    k_zero_edge_rows<<<1, 128, 0, stream>>>((float*)d_out, N, OUTC);
}
